// MixLinear_FP8GEMM_27814208209059
// MI455X (gfx1250) — 8-bit matrix operand order repaired and checked by reading, compiled, not run on hardware
//
#include <hip/hip_runtime.h>
#include <hip/hip_fp16.h>
#include <stdint.h>

typedef __attribute__((ext_vector_type(8)))  float v8f;
typedef __attribute__((ext_vector_type(16))) int   v16i;

#define FP8_MAXV 448.0f

__device__ __forceinline__ unsigned hmax2_bits(unsigned u) {
    u &= 0x7FFF7FFFu;
    unsigned a = u >> 16, b = u & 0xFFFFu;
    return a > b ? a : b;
}

__device__ __forceinline__ float half_bits_to_float(unsigned bits) {
    union { unsigned short u; __half h; } cv;
    cv.u = (unsigned short)bits;
    return __half2float(cv.h);
}

__device__ __forceinline__ float2 unpack_h2(unsigned u) {
    union { unsigned u; __half2 h2; } cv;
    cv.u = u;
    return __half22float2(cv.h2);
}

__device__ __forceinline__ float clamp_fp8(float v) {
    return fminf(fmaxf(v, -FP8_MAXV), FP8_MAXV);
}

__global__ void k_init(unsigned* amax) {
    if (threadIdx.x == 0) *amax = 0u;
}

__global__ void k_amax(const uint4* __restrict__ x, int n8, unsigned* __restrict__ amax) {
    __shared__ unsigned red[256];
    unsigned local = 0;
    int stride = gridDim.x * blockDim.x;
    for (int i = blockIdx.x * blockDim.x + threadIdx.x; i < n8; i += stride) {
        uint4 v = x[i];
        unsigned a = hmax2_bits(v.x);
        unsigned b = hmax2_bits(v.y);
        unsigned c = hmax2_bits(v.z);
        unsigned d = hmax2_bits(v.w);
        a = a > b ? a : b;
        c = c > d ? c : d;
        a = a > c ? a : c;
        local = local > a ? local : a;
    }
    red[threadIdx.x] = local;
    __syncthreads();
    for (int s = 128; s > 0; s >>= 1) {
        if (threadIdx.x < s) {
            unsigned o = red[threadIdx.x + s];
            if (o > red[threadIdx.x]) red[threadIdx.x] = o;
        }
        __syncthreads();
    }
    if (threadIdx.x == 0) atomicMax(amax, red[0]);
}

__global__ void k_quant(const uint4* __restrict__ src, uint2* __restrict__ dst, int n8,
                        const unsigned* __restrict__ amax_bits, int dynamic) {
    float inv = 1.0f;
    if (dynamic) {
        float amax  = half_bits_to_float(*amax_bits);
        float scale = fmaxf(amax / FP8_MAXV, 1e-12f);
        inv = 1.0f / scale;
    }
    int stride = gridDim.x * blockDim.x;
    for (int i = blockIdx.x * blockDim.x + threadIdx.x; i < n8; i += stride) {
        uint4 v = src[i];
        float2 f01 = unpack_h2(v.x);
        float2 f23 = unpack_h2(v.y);
        float2 f45 = unpack_h2(v.z);
        float2 f67 = unpack_h2(v.w);
        float q0 = clamp_fp8(f01.x * inv), q1 = clamp_fp8(f01.y * inv);
        float q2 = clamp_fp8(f23.x * inv), q3 = clamp_fp8(f23.y * inv);
        float q4 = clamp_fp8(f45.x * inv), q5 = clamp_fp8(f45.y * inv);
        float q6 = clamp_fp8(f67.x * inv), q7 = clamp_fp8(f67.y * inv);
        int lo = __builtin_amdgcn_cvt_pk_fp8_f32(q0, q1, 0,  false);
        lo     = __builtin_amdgcn_cvt_pk_fp8_f32(q2, q3, lo, true);
        int hi = __builtin_amdgcn_cvt_pk_fp8_f32(q4, q5, 0,  false);
        hi     = __builtin_amdgcn_cvt_pk_fp8_f32(q6, q7, hi, true);
        dst[i] = make_uint2((unsigned)lo, (unsigned)hi);
    }
}

__global__ __launch_bounds__(256) void k_gemm(
    const uint8_t* __restrict__ qx, const uint8_t* __restrict__ qw,
    const __half* __restrict__ bias, __half* __restrict__ out,
    const unsigned* __restrict__ amax_bits, int M, int N, int K)
{
    const int lane = threadIdx.x & 31;
    const int wave = threadIdx.x >> 5;
    const int wm   = wave >> 1;
    const int wn   = wave & 1;
    const int m0   = blockIdx.y * 256 + wm * 64;
    const int n0   = blockIdx.x * 128 + wn * 64;
    const int ml   = lane & 15;
    const int h    = lane >> 4;

    v8f acc[4][4];
    #pragma unroll
    for (int i = 0; i < 4; ++i)
        #pragma unroll
        for (int j = 0; j < 4; ++j)
            acc[i][j] = {};

    const unsigned long long* pa[4];
    #pragma unroll
    for (int i = 0; i < 4; ++i)
        pa[i] = reinterpret_cast<const unsigned long long*>(
            qx + (size_t)(m0 + i * 16 + ml) * K + h * 16);

    const int4* pb[4];
    #pragma unroll
    for (int j = 0; j < 4; ++j)
        pb[j] = reinterpret_cast<const int4*>(
            qw + (size_t)(n0 + j * 16 + ml) * K + h * 16);

    for (int kk = 0; kk < K; kk += 128) {
        union { v16i v; unsigned long long d[8]; } a[4];
        #pragma unroll
        for (int i = 0; i < 4; ++i) {
            const unsigned long long* p = pa[i] + (kk >> 3);
            #pragma unroll
            for (int j = 0; j < 8; ++j) a[i].d[j] = p[4 * (j >> 1) + (j & 1)];
        }
        union { v16i v; int4 q[4]; } b[4];
        #pragma unroll
        for (int j = 0; j < 4; ++j) {
            const int4* p = pb[j] + (kk >> 4);
            #pragma unroll
            for (int g = 0; g < 4; ++g) b[j].q[g] = p[2 * g];
        }
        #pragma unroll
        for (int i = 0; i < 4; ++i)
            #pragma unroll
            for (int j = 0; j < 4; ++j)
                acc[i][j] = __builtin_amdgcn_wmma_f32_16x16x128_fp8_fp8(
                    a[i].v, b[j].v, (short)0, acc[i][j], false, false);
    }

    float amax  = half_bits_to_float(*amax_bits);
    float scale = fmaxf(amax / FP8_MAXV, 1e-12f);

    #pragma unroll
    for (int j = 0; j < 4; ++j) {
        int col = n0 + j * 16 + ml;
        float bj = __half2float(bias[col]);
        #pragma unroll
        for (int i = 0; i < 4; ++i) {
            #pragma unroll
            for (int r = 0; r < 8; ++r) {
                int row = m0 + i * 16 + 8 * h + r;
                out[(size_t)row * N + col] = __float2half(acc[i][j][r] * scale + bj);
            }
        }
    }
}

extern "C" void kernel_launch(void* const* d_in, const int* in_sizes, int n_in,
                              void* d_out, int out_size, void* d_ws, size_t ws_size,
                              hipStream_t stream) {
    const __half* x    = (const __half*)d_in[0];
    const __half* w    = (const __half*)d_in[1];
    const __half* bias = (const __half*)d_in[2];
    __half* out        = (__half*)d_out;

    const int N = in_sizes[2];
    const int K = in_sizes[1] / N;
    const int M = in_sizes[0] / K;

    unsigned char* ws = (unsigned char*)d_ws;
    unsigned* amax = (unsigned*)ws;
    uint8_t*  qx   = ws + 256;
    uint8_t*  qw   = qx + (size_t)M * K;

    k_init<<<1, 1, 0, stream>>>(amax);
    k_amax<<<2048, 256, 0, stream>>>((const uint4*)x, M * K / 8, amax);
    k_quant<<<4096, 256, 0, stream>>>((const uint4*)x, (uint2*)qx, M * K / 8, amax, 1);
    k_quant<<<1024, 256, 0, stream>>>((const uint4*)w, (uint2*)qw, N * K / 8, amax, 0);

    dim3 grid(N / 128, M / 256);
    k_gemm<<<grid, 256, 0, stream>>>(qx, qw, bias, out, amax, M, N, K);
}
